// DecGridDeepVPN_33328946217237
// MI455X (gfx1250) — matrix operand order repaired and checked by reading, not compiled, not run on hardware
//
#include <hip/hip_runtime.h>

typedef __attribute__((ext_vector_type(16))) _Float16 v16h;
typedef __attribute__((ext_vector_type(8)))  _Float16 v8h;
typedef __attribute__((ext_vector_type(8)))  float    v8f;

#define S_NUM 81
#define N_AG  20
#define N_ACT 5
#define TB    32
#define OBS_STRIDE 182
#define AC_STRIDE  405
#define ROWS (TB * N_AG)      // 640 MLP rows per block
#define H0_STRIDE 40          // padded f16 stride (80B = 16B-multiple)
#define K_PAD 104             // 81 padded to multiple of 32-ish, 16B-aligned rows

struct Smem {
  _Float16 nsc[TB][K_PAD];        // f16 next_state_count, zero-padded K
  _Float16 w0a_t[32][K_PAD];      // w0a_t[n][k] = W0[k][n], k<81 else 0
  float    base[TB][32];          // nsc @ W0a
  float    w0b[S_NUM][32];        // one-hot(loc) rows
  float    w0c[N_ACT][32];        // local_actions rows
  float    w0d[N_AG][32];         // one-hot(agent) rows
  _Float16 w1t[32][32];           // w1t[n][k] = W1[k][n]
  float    w2v[32];
  float    la[TB * N_AG * N_ACT];
  int      loc[TB * N_AG];
  float    reward[TB];
  _Float16 h0[ROWS][H0_STRIDE];   // relu(layer1) in f16
};

__device__ __forceinline__ v16h load_frag(const _Float16* p0, const _Float16* p1) {
  v8h lo = *(const v8h*)p0;
  v8h hi = *(const v8h*)p1;
  v16h r;
#pragma unroll
  for (int i = 0; i < 8; ++i) { r[i] = lo[i]; r[8 + i] = hi[i]; }
  return r;
}

__global__ __launch_bounds__(256, 2)
void vpn_fused(const float* __restrict__ obs,
               const float* __restrict__ action_count,
               const float* __restrict__ local_actions,
               const float* __restrict__ W0,
               const float* __restrict__ W1,
               const float* __restrict__ W2,
               const float* __restrict__ b2,
               float* __restrict__ out)
{
  extern __shared__ __align__(16) char smem_raw[];
  Smem& sm = *reinterpret_cast<Smem*>(smem_raw);
  const int tid  = threadIdx.x;
  const int lane = tid & 31;
  const int wave = tid >> 5;
  const int b0   = blockIdx.x * TB;

  if (tid < TB) sm.reward[tid] = 0.f;
  __syncthreads();

  // ---------------- phase 0: nsc gather + reward + weight staging ----------
  {
    const int dr_[4]  = {-1, 1, 0, 0};
    const int dc_[4]  = {0, 0, -1, 1};
    const int rev_[4] = {2, 1, 4, 3};   // OOB self-loop action for each reverse move
    for (int idx = tid; idx < TB * S_NUM; idx += 256) {
      int bl = idx / S_NUM;
      int d  = idx - bl * S_NUM;
      int r  = d / 9, c = d - r * 9;
      const float* ac = action_count + (size_t)(b0 + bl) * AC_STRIDE;
      __builtin_prefetch(ac, 0, 0);
      float sum = ac[d * 5];            // "stay" inflow
#pragma unroll
      for (int a = 1; a <= 4; ++a) {
        int sr = r - dr_[a - 1], sc = c - dc_[a - 1];
        int srcflat = (sr >= 0 && sr < 9 && sc >= 0 && sc < 9)
                      ? ((sr * 9 + sc) * 5 + a)
                      : (d * 5 + rev_[a - 1]);   // OOB self-loop
        sum += ac[srcflat];
      }
      float demand = obs[(size_t)(b0 + bl) * OBS_STRIDE + S_NUM + d];
      sm.nsc[bl][d] = (_Float16)sum;
      atomicAdd(&sm.reward[bl], fminf(sum, demand));   // ds_add_f32
    }
    for (int idx = tid; idx < TB * (K_PAD - S_NUM); idx += 256) {
      int bl = idx / (K_PAD - S_NUM);
      sm.nsc[bl][S_NUM + (idx - bl * (K_PAD - S_NUM))] = (_Float16)0.f;
    }
    for (int idx = tid; idx < 32 * K_PAD; idx += 256) {
      int n = idx / K_PAD, k = idx - n * K_PAD;
      sm.w0a_t[n][k] = (_Float16)((k < S_NUM) ? W0[k * 32 + n] : 0.f);
    }
    for (int idx = tid; idx < S_NUM * 32; idx += 256)
      sm.w0b[0][idx] = W0[S_NUM * 32 + idx];
    for (int idx = tid; idx < N_ACT * 32; idx += 256)
      sm.w0c[0][idx] = W0[2 * S_NUM * 32 + idx];
    for (int idx = tid; idx < N_AG * 32; idx += 256)
      sm.w0d[0][idx] = W0[(2 * S_NUM + N_ACT) * 32 + idx];
    for (int idx = tid; idx < 32 * 32; idx += 256) {
      int n = idx >> 5, k = idx & 31;
      sm.w1t[n][k] = (_Float16)W1[k * 32 + n];
    }
    if (tid < 32) sm.w2v[tid] = W2[tid];
    for (int idx = tid; idx < TB * N_AG * N_ACT; idx += 256)
      sm.la[idx] = local_actions[(size_t)b0 * (N_AG * N_ACT) + idx];
    for (int idx = tid; idx < TB * N_AG; idx += 256) {
      int bl = idx / N_AG, n = idx - bl * N_AG;
      int l = (int)obs[(size_t)(b0 + bl) * OBS_STRIDE + 2 * S_NUM + n];
      sm.loc[idx] = (l < 0) ? 0 : (l >= S_NUM ? S_NUM - 1 : l);
    }
  }
  __syncthreads();

  // ---------------- phase 1: base = nsc @ W0a   (WMMA, waves 0-1) ----------
  if (wave < 2) {
    v8f c0 = {}, c1 = {};
    const int arow = lane & 15;
    const int ksel = (lane < 16) ? 0 : 16;   // A-frag K sub-base per half-wave
    const int kbse = (lane < 16) ? 0 : 16;   // B-frag K sub-base per half-wave
    const int coln = lane & 15;
#pragma unroll
    for (int kt = 0; kt < 3; ++kt) {         // K = 96 in three 32-steps
      int k0 = kt * 32 + ksel;
      v16h a = load_frag(&sm.nsc[wave * 16 + arow][k0],
                         &sm.nsc[wave * 16 + arow][k0 + 8]);
      int kb = kt * 32 + kbse;
      v16h bf0 = load_frag(&sm.w0a_t[coln][kb],      &sm.w0a_t[coln][kb + 8]);
      v16h bf1 = load_frag(&sm.w0a_t[16 + coln][kb], &sm.w0a_t[16 + coln][kb + 8]);
      c0 = __builtin_amdgcn_wmma_f32_16x16x32_f16(false, a, false, bf0, (short)0, c0, false, false);
      c1 = __builtin_amdgcn_wmma_f32_16x16x32_f16(false, a, false, bf1, (short)0, c1, false, false);
    }
    int rbase = wave * 16 + ((lane < 16) ? 0 : 8);
#pragma unroll
    for (int j = 0; j < 8; ++j) {
      sm.base[rbase + j][coln]      = c0[j];
      sm.base[rbase + j][16 + coln] = c1[j];
    }
  }
  __syncthreads();

  // ---------------- phase 2: assemble h0 = relu(layer1) --------------------
  for (int idx = tid; idx < ROWS * 32; idx += 256) {
    int row = idx >> 5, k = idx & 31;
    int bl = row / N_AG;
    int ag = row - bl * N_AG;
    const float* lap = &sm.la[row * N_ACT];
    int l = sm.loc[row];
    float v = sm.base[bl][k] + sm.w0b[l][k] + sm.w0d[ag][k];
#pragma unroll
    for (int a2 = 0; a2 < N_ACT; ++a2) v += lap[a2] * sm.w0c[a2][k];
    sm.h0[row][k] = (_Float16)fmaxf(v, 0.f);
  }
  __syncthreads();

  // ---------------- phase 3: layer2 WMMA + layer3 dot + store --------------
  {
    const int coln = lane & 15;
    const int kbse = (lane < 16) ? 0 : 16;
    const int ksel = (lane < 16) ? 0 : 16;
    const int arow = lane & 15;
    v16h bf0 = load_frag(&sm.w1t[coln][kbse],      &sm.w1t[coln][kbse + 8]);
    v16h bf1 = load_frag(&sm.w1t[16 + coln][kbse], &sm.w1t[16 + coln][kbse + 8]);
    float w2a = sm.w2v[coln];
    float w2b = sm.w2v[16 + coln];
    float b2v = b2[0];
#pragma unroll
    for (int i = 0; i < 5; ++i) {            // 40 tiles / 8 waves
      int t = wave * 5 + i;
      v16h a = load_frag(&sm.h0[t * 16 + arow][ksel],
                         &sm.h0[t * 16 + arow][ksel + 8]);
      v8f c0 = {}, c1 = {};
      c0 = __builtin_amdgcn_wmma_f32_16x16x32_f16(false, a, false, bf0, (short)0, c0, false, false);
      c1 = __builtin_amdgcn_wmma_f32_16x16x32_f16(false, a, false, bf1, (short)0, c1, false, false);
      float p[8];
#pragma unroll
      for (int j = 0; j < 8; ++j)
        p[j] = fmaxf(c0[j], 0.f) * w2a + fmaxf(c1[j], 0.f) * w2b;
#pragma unroll
      for (int off = 1; off < 16; off <<= 1) {
#pragma unroll
        for (int j = 0; j < 8; ++j) p[j] += __shfl_xor(p[j], off, 32);
      }
      if ((lane & 8) == 0) {                 // lanes 0-7 & 16-23 write 16 rows
        int j = lane & 7;
        float psel = p[0];
#pragma unroll
        for (int jj = 1; jj < 8; ++jj) if (j == jj) psel = p[jj];
        int m  = t * 16 + ((lane < 16) ? 0 : 8) + j;
        int bl = m / N_AG;
        int ag = m - bl * N_AG;
        out[(size_t)(b0 + bl) * N_AG + ag] = psel + sm.reward[bl] + b2v;
      }
    }
  }
}

extern "C" void kernel_launch(void* const* d_in, const int* in_sizes, int n_in,
                              void* d_out, int out_size, void* d_ws, size_t ws_size,
                              hipStream_t stream) {
  const float* obs           = (const float*)d_in[0];
  const float* action_count  = (const float*)d_in[1];
  const float* local_actions = (const float*)d_in[2];
  const float* W0            = (const float*)d_in[3];
  const float* W1            = (const float*)d_in[4];
  const float* W2            = (const float*)d_in[5];
  const float* b2            = (const float*)d_in[6];
  float* out = (float*)d_out;
  int batch  = in_sizes[0] / OBS_STRIDE;   // 32768
  int blocks = batch / TB;                 // 1024
  hipLaunchKernelGGL(vpn_fused, dim3(blocks), dim3(256), sizeof(Smem), stream,
                     obs, action_count, local_actions, W0, W1, W2, b2, out);
}
